// H3TCSNetwork_16518444220506
// MI455X (gfx1250) — matrix operand order repaired and checked by reading, not compiled, not run on hardware
//
#include <hip/hip_runtime.h>
#include <stdint.h>

typedef __attribute__((ext_vector_type(16))) _Float16 v16h;
typedef __attribute__((ext_vector_type(8)))  float    v8f;
typedef __attribute__((ext_vector_type(4)))  uint32_t v4u;

#define HID    256
#define NOUT   2695      // 77 heads * 35
#define OUTH   35
#define ROWS   32        // two 16-row M-tiles per block
#define LSTR   264       // padded LDS row stride in halves (256 + 8)
#define NTILES 169       // ceil(2695 / 16)

union AV { v16h v; v4u q[2]; };

// A fragment (16x32 f16, M on lanes, K striped over VGPRs) from LDS activations.
// Lanes 0-15: M=lane; VGPR0-3 hold K=0..7 (+8 for lanes 16-31), VGPR4-7 hold
// K=16..23 (+8 for lanes 16-31). Halves 0..7 / 16..23 contiguous -> 2x ds_load_b128.
__device__ __forceinline__ v16h ldsA(const _Float16* base, int lane, int k0) {
  int m  = lane & 15;
  int kh = k0 + ((lane >> 4) << 3);
  const _Float16* p = base + m * LSTR + kh;
  AV a;
  a.q[0] = *(const v4u*)(p);
  a.q[1] = *(const v4u*)(p + 16);
  return a.v;
}

// B fragment (32x16 f16, N on lanes, K striped over VGPRs) from transposed
// weights Wt[N][K] (K contiguous). Same K order as A: halves 0..7 hold K=0..7
// (+8 for lanes 16-31), halves 8..15 hold K=16..23 (+8) -> 2x global_load_b128.
__device__ __forceinline__ v16h gB(const _Float16* __restrict__ Wt, int row,
                                   int k0, int lane) {
  int k = k0 + ((lane >> 4) << 3);
  const _Float16* p = Wt + (size_t)row * HID + k;
  AV b;
  b.q[0] = *(const v4u*)(p);
  b.q[1] = *(const v4u*)(p + 16);
  return b.v;
}

__device__ __forceinline__ float silu(float a) {
  return a / (1.0f + __expf(-a));
}

// Convert f32 weights (K x N, row-major) to f16 transposed (N x K).
__global__ void cvt_transpose(const float* __restrict__ in,
                              _Float16* __restrict__ out, int K, int N) {
  int idx = blockIdx.x * 256 + threadIdx.x;
  if (idx >= K * N) return;
  int n = idx / K;
  int k = idx - n * K;
  out[idx] = (_Float16)in[(size_t)k * N + n];
}

__global__ __launch_bounds__(256) void h3_fused(
    const float* __restrict__ x,  const float* __restrict__ W0,
    const float* __restrict__ b0, const float* __restrict__ b1,
    const float* __restrict__ b2, const float* __restrict__ b3,
    const float* __restrict__ bh,
    const _Float16* __restrict__ W1t, const _Float16* __restrict__ W2t,
    const _Float16* __restrict__ W3t, const _Float16* __restrict__ Wht,
    float* __restrict__ out) {
  __shared__ _Float16 act0[ROWS * LSTR];
  __shared__ _Float16 act1[ROWS * LSTR];
  __shared__ float xs[ROWS * 8];
  __shared__ float prof[ROWS * 4];   // [unused, left, right, neck] per row

  const int tid     = threadIdx.x;
  const int lane    = tid & 31;
  const int wave    = tid >> 5;
  const int rowBase = blockIdx.x * ROWS;

  // ---- load x tile ----
  if (tid < ROWS * 7) {
    int m = tid / 7, j = tid - m * 7;
    xs[m * 8 + j] = x[(size_t)(rowBase + m) * 7 + j];
  }
  __syncthreads();

  // ---- per-row profiles (lam = x[:,0]) ----
  if (tid < ROWS) {
    float lam = xs[tid * 8];
    float s   = 1.0f / (1.0f + __expf(-lam * (5.0f / 0.15f)));
    prof[tid * 4 + 0] = 1.0f;
    prof[tid * 4 + 1] = 1.0f - s;                            // left
    prof[tid * 4 + 2] = s;                                   // right
    prof[tid * 4 + 3] = __expf(-lam * lam * (1.0f / 0.04f)); // neck
  }

  // ---- layer 0 (K=7): one thread per output column, all 32 rows ----
  {
    int n = tid;
    float w[7];
#pragma unroll
    for (int j = 0; j < 7; ++j) w[j] = W0[j * HID + n];
    float bb = b0[n];
#pragma unroll
    for (int m = 0; m < ROWS; ++m) {
      float a = bb;
#pragma unroll
      for (int j = 0; j < 7; ++j) a = fmaf(xs[m * 8 + j], w[j], a);
      act0[m * LSTR + n] = (_Float16)silu(a);
    }
  }
  __syncthreads();

  // ---- layers 1..3: 256x256 GEMM + bias + SiLU ----
  // Each wave: 2 N-tiles x 2 M-tiles; B fragments reused across M-tiles.
#pragma unroll
  for (int L = 0; L < 3; ++L) {
    const _Float16* Wt  = (L == 0) ? W1t : (L == 1) ? W2t : W3t;
    const float*    bsL = (L == 0) ? b1  : (L == 1) ? b2  : b3;
    const _Float16* src = (L & 1) ? act1 : act0;
    _Float16*       dst = (L & 1) ? act0 : act1;

    const int n0  = wave * 32;
    const int nlo = lane & 15;
    const int rA  = n0 + nlo;
    const int rB  = n0 + 16 + nlo;
    v8f acc00 = {}, acc01 = {}, acc10 = {}, acc11 = {};
#pragma unroll
    for (int kt = 0; kt < 8; ++kt) {
      v16h A0 = ldsA(src,             lane, kt * 32);   // rows 0..15
      v16h A1 = ldsA(src + 16 * LSTR, lane, kt * 32);   // rows 16..31
      v16h B0 = gB(Wt, rA, kt * 32, lane);
      v16h B1 = gB(Wt, rB, kt * 32, lane);
      acc00 = __builtin_amdgcn_wmma_f32_16x16x32_f16(false, A0, false, B0,
                                                     (short)0, acc00, false, false);
      acc10 = __builtin_amdgcn_wmma_f32_16x16x32_f16(false, A1, false, B0,
                                                     (short)0, acc10, false, false);
      acc01 = __builtin_amdgcn_wmma_f32_16x16x32_f16(false, A0, false, B1,
                                                     (short)0, acc01, false, false);
      acc11 = __builtin_amdgcn_wmma_f32_16x16x32_f16(false, A1, false, B1,
                                                     (short)0, acc11, false, false);
    }
    float bb0 = bsL[rA];
    float bb1 = bsL[rB];
#pragma unroll
    for (int v = 0; v < 8; ++v) {
      int m = v + ((lane >> 4) << 3);
      dst[m        * LSTR + n0 + nlo]      = (_Float16)silu(acc00[v] + bb0);
      dst[(m + 16) * LSTR + n0 + nlo]      = (_Float16)silu(acc10[v] + bb0);
      dst[m        * LSTR + n0 + 16 + nlo] = (_Float16)silu(acc01[v] + bb1);
      dst[(m + 16) * LSTR + n0 + 16 + nlo] = (_Float16)silu(acc11[v] + bb1);
    }
    __syncthreads();
  }

  // ---- head: 256 -> 2695, per-head scaling; A resident in registers ----
  const _Float16* fsrc = act1;   // after L=2 the result lives in act1
  v16h Ah0[8], Ah1[8];
#pragma unroll
  for (int kt = 0; kt < 8; ++kt) {
    Ah0[kt] = ldsA(fsrc,             lane, kt * 32);
    Ah1[kt] = ldsA(fsrc + 16 * LSTR, lane, kt * 32);
  }

  const int nlo = lane & 15;
  for (int t = wave; t < NTILES; t += 8) {
    int c  = t * 16 + nlo;
    int cc = (c < NOUT) ? c : (NOUT - 1);   // clamp loads on last tile
    v8f acc0 = {}, acc1 = {};
#pragma unroll
    for (int kt = 0; kt < 8; ++kt) {
      v16h Bm = gB(Wht, cc, kt * 32, lane);
      acc0 = __builtin_amdgcn_wmma_f32_16x16x32_f16(false, Ah0[kt], false, Bm,
                                                    (short)0, acc0, false, false);
      acc1 = __builtin_amdgcn_wmma_f32_16x16x32_f16(false, Ah1[kt], false, Bm,
                                                    (short)0, acc1, false, false);
    }
    float bias = bh[cc];
    int h = cc / OUTH;
#pragma unroll
    for (int v = 0; v < 8; ++v) {
      int m = v + ((lane >> 4) << 3);
      float sc0 = (h < 35) ? 1.0f
                : (h < 49) ? prof[m * 4 + 1]
                : (h < 63) ? prof[m * 4 + 2]
                :            prof[m * 4 + 3];
      float sc1 = (h < 35) ? 1.0f
                : (h < 49) ? prof[(m + 16) * 4 + 1]
                : (h < 63) ? prof[(m + 16) * 4 + 2]
                :            prof[(m + 16) * 4 + 3];
      if (c < NOUT) {
        out[(size_t)(rowBase + m)      * NOUT + c] = (acc0[v] + bias) * sc0;
        out[(size_t)(rowBase + m + 16) * NOUT + c] = (acc1[v] + bias) * sc1;
      }
    }
  }
}

extern "C" void kernel_launch(void* const* d_in, const int* in_sizes, int n_in,
                              void* d_out, int out_size, void* d_ws, size_t ws_size,
                              hipStream_t stream) {
  const float* x  = (const float*)d_in[0];
  const float* W0 = (const float*)d_in[1];
  const float* b0 = (const float*)d_in[2];
  const float* W1 = (const float*)d_in[3];
  const float* b1 = (const float*)d_in[4];
  const float* W2 = (const float*)d_in[5];
  const float* b2 = (const float*)d_in[6];
  const float* W3 = (const float*)d_in[7];
  const float* b3 = (const float*)d_in[8];
  const float* Wh = (const float*)d_in[9];
  const float* bh = (const float*)d_in[10];
  float* out = (float*)d_out;

  _Float16* W1t = (_Float16*)d_ws;
  _Float16* W2t = W1t + HID * HID;
  _Float16* W3t = W2t + HID * HID;
  _Float16* Wht = W3t + HID * HID;   // total ws use: ~1.73 MB

  cvt_transpose<<<(HID * HID + 255) / 256, 256, 0, stream>>>(W1, W1t, HID, HID);
  cvt_transpose<<<(HID * HID + 255) / 256, 256, 0, stream>>>(W2, W2t, HID, HID);
  cvt_transpose<<<(HID * HID + 255) / 256, 256, 0, stream>>>(W3, W3t, HID, HID);
  cvt_transpose<<<(HID * NOUT + 255) / 256, 256, 0, stream>>>(Wh, Wht, HID, NOUT);

  int B = in_sizes[0] / 7;
  h3_fused<<<B / ROWS, 256, 0, stream>>>(x, W0, b0, b1, b2, b3, bh,
                                         W1t, W2t, W3t, Wht, out);
}
